// SwinTransformerBlock_34179349741713
// MI455X (gfx1250) — compile-verified, matrix operand order checked by reading, not run on hardware
//
#include <hip/hip_runtime.h>
#include <hip/hip_bf16.h>
#include <math.h>

typedef __bf16 bf16;
typedef __bf16 v16bf __attribute__((ext_vector_type(16)));
typedef float  v8f   __attribute__((ext_vector_type(8)));

#define WMMA_BF16(a, b, c) \
  __builtin_amdgcn_wmma_f32_16x16x32_bf16(false, (a), false, (b), (short)0, (c), false, false)

// ---- problem constants ----
constexpr int  BB   = 32;          // batch
constexpr int  HW   = 64;          // H = W
constexpr int  C_   = 192;
constexpr int  NH_  = 6;
constexpr int  HD   = 32;          // head dim
constexpr int  SSz  = 4;           // shift
constexpr int  Ntok = 64;          // tokens per 8x8 window
constexpr int  BW   = BB * 64;     // 2048 windows total
constexpr long T    = (long)BB * HW * HW;   // 131072 tokens
constexpr int  HID  = 4 * C_;      // 768

// ---------------------------------------------------------------------------
// WMMA operand loaders (CDNA5 16-bit A layout, ISA 7.12.2).
// Per lane the 16 bf16 elements are two contiguous 16B runs -> the compiler
// lowers each call to 2x global_load_b128 / ds_load_b128.
// ---------------------------------------------------------------------------
__device__ inline v16bf load_a(const bf16* base, int ld) {
  int lane = threadIdx.x & 31;
  int half = lane >> 4, r = lane & 15;
  const bf16* p = base + (long)r * ld + (half ? 8 : 0);
  v16bf o;
#pragma unroll
  for (int j = 0; j < 8; ++j) {
    int k = (j < 4) ? (2 * j) : (8 + 2 * j);
    o[2 * j]     = p[k];
    o[2 * j + 1] = p[k + 1];
  }
  return o;
}

// B tile 32x16 from row-major [K, N] memory (B stored K-major), row stride ld.
__device__ inline v16bf load_b_kn(const bf16* base, int ld) {
  int lane = threadIdx.x & 31;
  int half = lane >> 4, n = lane & 15;
  v16bf o;
#pragma unroll
  for (int j = 0; j < 8; ++j) {
    int kk = ((j < 4) ? (2 * j) : (8 + 2 * j)) + (half ? 8 : 0);
    o[2 * j]     = base[(long)kk * ld + n];
    o[2 * j + 1] = base[(long)(kk + 1) * ld + n];
  }
  return o;
}

// ---------------------------------------------------------------------------
// K0: convert all weight matrices fp32 -> bf16 (tiny; L2-resident afterwards)
// ---------------------------------------------------------------------------
__global__ void k_cvt(const float* __restrict__ s0, const float* __restrict__ s1,
                      const float* __restrict__ s2, const float* __restrict__ s3,
                      bf16* __restrict__ d0, bf16* __restrict__ d1,
                      bf16* __restrict__ d2, bf16* __restrict__ d3) {
  int i = blockIdx.x * 256 + threadIdx.x;
  if (i < 576 * 192) d0[i] = (bf16)s0[i];
  if (i < 192 * 192) d1[i] = (bf16)s1[i];
  if (i < 768 * 192) d2[i] = (bf16)s2[i];
  if (i < 192 * 768) d3[i] = (bf16)s3[i];
}

// ---------------------------------------------------------------------------
// K1: LN1 + roll(-shift) + window partition, fp32 -> bf16.
// One wave per destination token (window order). 6 channels per lane.
// ---------------------------------------------------------------------------
__global__ __launch_bounds__(256) void k_ln1(const float* __restrict__ x,
                                             const float* __restrict__ g,
                                             const float* __restrict__ be,
                                             bf16* __restrict__ xw) {
  int  wave = threadIdx.x >> 5, lane = threadIdx.x & 31;
  long rowW = (long)blockIdx.x * 8 + wave;
  int  n    = (int)(rowW & 63);
  long widx = rowW >> 6;
  int  ww   = (int)(widx & 7), wh = (int)((widx >> 3) & 7);
  long b    = widx >> 6;
  int  hs   = (wh * 8 + (n >> 3) + SSz) & 63;
  int  wsr  = (ww * 8 + (n & 7) + SSz) & 63;
  const float* src = x + ((b * 64 + hs) * 64 + wsr) * (long)C_;

  float vals[6], s = 0.f, ss = 0.f;
#pragma unroll
  for (int j = 0; j < 6; ++j) {
    float t = src[j * 32 + lane];
    vals[j] = t; s += t; ss += t * t;
  }
#pragma unroll
  for (int m = 16; m >= 1; m >>= 1) {
    s  += __shfl_xor(s, m, 32);
    ss += __shfl_xor(ss, m, 32);
  }
  float mu  = s * (1.f / 192.f);
  float var = ss * (1.f / 192.f) - mu * mu;
  float inv = rsqrtf(var + 1e-5f);
  bf16* dst = xw + rowW * C_;
#pragma unroll
  for (int j = 0; j < 6; ++j) {
    int ch = j * 32 + lane;
    dst[ch] = (bf16)((vals[j] - mu) * inv * g[ch] + be[ch]);
  }
}

// ---------------------------------------------------------------------------
// K2: QKV GEMM [T,192] x [576,192]^T. Each wave owns a 16x64 strip (4 accs),
// so one A-tile load feeds 4 WMMAs. Fused bias + q-scale, scatter into
// [win, head, 64, 32] bf16.
// ---------------------------------------------------------------------------
__global__ __launch_bounds__(256) void k_qkv(const bf16* __restrict__ xw,
                                             const bf16* __restrict__ wq,
                                             const float* __restrict__ qkv_b,
                                             bf16* __restrict__ q,
                                             bf16* __restrict__ k,
                                             bf16* __restrict__ v) {
  int  wave = threadIdx.x >> 5;
  long job = (long)blockIdx.x * 8 + wave;       // 8192 * 9 jobs
  int  mt = (int)(job / 9), ng = (int)(job % 9);
  int  m0 = mt * 16, n0 = ng * 64;

  const bf16* arow = xw + (long)m0 * C_;
  const bf16* brow = wq + (long)n0 * C_;
  __builtin_prefetch(arow + 16 * C_, 0, 0);     // next M strip of the stream

  v8f acc[4] = {};
#pragma unroll
  for (int kk = 0; kk < C_; kk += 32) {
    v16bf a = load_a(arow + kk, C_);
#pragma unroll
    for (int s = 0; s < 4; ++s)
      acc[s] = WMMA_BF16(a, load_a(brow + (long)s * 16 * C_ + kk, C_), acc[s]);
  }

  int lane = threadIdx.x & 31, half = lane >> 4, c16 = lane & 15;
#pragma unroll
  for (int s = 0; s < 4; ++s) {
    int   col = n0 + s * 16 + c16;
    int   which = col / 192, rem = col % 192;
    int   head = rem / 32, d = rem % 32;
    float bias  = qkv_b[col];
    float scale = (which == 0) ? 0.17677669529663687f : 1.0f;  // hd^-0.5
    bf16* dst = (which == 0) ? q : (which == 1) ? k : v;
#pragma unroll
    for (int i = 0; i < 8; ++i) {
      int row  = m0 + i + (half ? 8 : 0);
      int widx = row >> 6, n = row & 63;
      dst[(((long)widx * NH_ + head) * Ntok + n) * HD + d] =
          (bf16)((acc[s][i] + bias) * scale);
    }
  }
}

// ---------------------------------------------------------------------------
// K3: attention per (window, head). 4 waves; each owns a 16x64 strip of S.
// Rel-pos bias and shift mask computed analytically. Softmax via 16-lane
// shfl_xor (each row lives in one lane-half of the C/D layout). P restaged
// through LDS into A-layout for P@V.
// ---------------------------------------------------------------------------
__global__ __launch_bounds__(128) void k_attn(const bf16* __restrict__ q,
                                              const bf16* __restrict__ kq,
                                              const bf16* __restrict__ v,
                                              const float* __restrict__ rpb,
                                              bf16* __restrict__ o) {
  __shared__ float srpb[225 * NH_];
  __shared__ bf16  P[4][16 * 64];
  int tid = threadIdx.x;
  for (int i = tid; i < 225 * NH_; i += 128) srpb[i] = rpb[i];
  __syncthreads();

  int head   = blockIdx.x % NH_;
  int w      = blockIdx.x / NH_;
  int wwin   = w & 63;                 // window within image
  int wimg_w = wwin & 7, wimg_h = (wwin >> 3) & 7;
  const bf16* qb = q  + ((long)w * NH_ + head) * Ntok * HD;
  const bf16* kb = kq + ((long)w * NH_ + head) * Ntok * HD;
  const bf16* vb = v  + ((long)w * NH_ + head) * Ntok * HD;

  int wave = tid >> 5, lane = tid & 31;
  int half = lane >> 4, c16 = lane & 15;
  int m0 = wave * 16;

  // S = (q*scale) @ k^T : 4 x 16x16 tiles, K = 32 = hd (one WMMA each)
  v16bf aq = load_a(qb + (long)m0 * HD, HD);
  v8f s[4];
#pragma unroll
  for (int t = 0; t < 4; ++t) {
    v8f z = {};
    s[t] = WMMA_BF16(aq, load_a(kb + (long)t * 16 * HD, HD), z);
  }

  // + relative position bias + shift mask (analytic)
#pragma unroll
  for (int t = 0; t < 4; ++t) {
    int mcol = t * 16 + c16;
    int r2 = mcol >> 3, c2 = mcol & 7;
    int gh2 = wimg_h * 8 + r2, gw2 = wimg_w * 8 + c2;
    int lab2 = ((gh2 < 56) ? 0 : (gh2 < 60 ? 1 : 2)) * 3 +
               ((gw2 < 56) ? 0 : (gw2 < 60 ? 1 : 2));
#pragma unroll
    for (int i = 0; i < 8; ++i) {
      int nrow = m0 + i + (half ? 8 : 0);
      int r1 = nrow >> 3, c1 = nrow & 7;
      int gh1 = wimg_h * 8 + r1, gw1 = wimg_w * 8 + c1;
      int lab1 = ((gh1 < 56) ? 0 : (gh1 < 60 ? 1 : 2)) * 3 +
                 ((gw1 < 56) ? 0 : (gw1 < 60 ? 1 : 2));
      int rpi = (r1 - r2 + 7) * 15 + (c1 - c2 + 7);
      s[t][i] += srpb[rpi * NH_ + head] + ((lab1 == lab2) ? 0.f : -100.f);
    }
  }

  // softmax over 64 keys per row; write P (bf16) into LDS row-major
#pragma unroll
  for (int i = 0; i < 8; ++i) {
    float mx = fmaxf(fmaxf(s[0][i], s[1][i]), fmaxf(s[2][i], s[3][i]));
#pragma unroll
    for (int m = 8; m >= 1; m >>= 1) mx = fmaxf(mx, __shfl_xor(mx, m, 32));
    float sum = 0.f;
#pragma unroll
    for (int t = 0; t < 4; ++t) { float e = __expf(s[t][i] - mx); s[t][i] = e; sum += e; }
#pragma unroll
    for (int m = 8; m >= 1; m >>= 1) sum += __shfl_xor(sum, m, 32);
    float inv = 1.f / sum;
#pragma unroll
    for (int t = 0; t < 4; ++t)
      P[wave][(i + (half ? 8 : 0)) * 64 + t * 16 + c16] = (bf16)(s[t][i] * inv);
  }

  // O = P @ V : 16x32 per wave = 2 tiles, K = 64 (2 WMMAs each)
  const bf16* PL = &P[wave][0];
#pragma unroll
  for (int d0 = 0; d0 < 32; d0 += 16) {
    v8f acc = {};
#pragma unroll
    for (int k0 = 0; k0 < 64; k0 += 32) {
      v16bf a = load_a(PL + k0, 64);
      v16bf b = load_b_kn(vb + (long)k0 * HD + d0, HD);
      acc = WMMA_BF16(a, b, acc);
    }
#pragma unroll
    for (int i = 0; i < 8; ++i) {
      int nrow = m0 + i + (half ? 8 : 0);
      o[((long)w * Ntok + nrow) * C_ + head * HD + d0 + c16] = (bf16)acc[i];
    }
  }
}

// ---------------------------------------------------------------------------
// K4: proj GEMM (16x64 strip per wave) + window-reverse + roll(+shift)
//     + residual -> h (fp32) written into d_out.
// ---------------------------------------------------------------------------
__global__ __launch_bounds__(256) void k_proj(const bf16* __restrict__ o,
                                              const bf16* __restrict__ wp,
                                              const float* __restrict__ pb,
                                              const float* __restrict__ x,
                                              float* __restrict__ h) {
  int  wave = threadIdx.x >> 5;
  long job = (long)blockIdx.x * 8 + wave;       // 8192 * 3 jobs
  int  mt = (int)(job / 3), ng = (int)(job % 3);
  int  m0 = mt * 16, n0 = ng * 64;

  const bf16* arow = o  + (long)m0 * C_;
  const bf16* brow = wp + (long)n0 * C_;
  __builtin_prefetch(arow + 16 * C_, 0, 0);

  v8f acc[4] = {};
#pragma unroll
  for (int kk = 0; kk < C_; kk += 32) {
    v16bf a = load_a(arow + kk, C_);
#pragma unroll
    for (int s = 0; s < 4; ++s)
      acc[s] = WMMA_BF16(a, load_a(brow + (long)s * 16 * C_ + kk, C_), acc[s]);
  }

  int lane = threadIdx.x & 31, half = lane >> 4, c16 = lane & 15;
#pragma unroll
  for (int s = 0; s < 4; ++s) {
    int   col = n0 + s * 16 + c16;
    float bias = pb[col];
#pragma unroll
    for (int i = 0; i < 8; ++i) {
      long rowW = m0 + i + (half ? 8 : 0);
      int  n = (int)(rowW & 63);
      long widx = rowW >> 6;
      int  ww = (int)(widx & 7), wh = (int)((widx >> 3) & 7);
      long b  = widx >> 6;
      int  hf = (wh * 8 + (n >> 3) + SSz) & 63;   // un-shift
      int  wf = (ww * 8 + (n & 7) + SSz) & 63;
      long flat = (b * 64 + hf) * 64 + wf;
      h[flat * C_ + col] = x[flat * C_ + col] + acc[s][i] + bias;
    }
  }
}

// ---------------------------------------------------------------------------
// K5: LN2 (natural token order), fp32 -> bf16
// ---------------------------------------------------------------------------
__global__ __launch_bounds__(256) void k_ln2(const float* __restrict__ h,
                                             const float* __restrict__ g,
                                             const float* __restrict__ be,
                                             bf16* __restrict__ y) {
  int  wave = threadIdx.x >> 5, lane = threadIdx.x & 31;
  long tok = (long)blockIdx.x * 8 + wave;
  const float* src = h + tok * C_;
  float vals[6], s = 0.f, ss = 0.f;
#pragma unroll
  for (int j = 0; j < 6; ++j) {
    float t = src[j * 32 + lane];
    vals[j] = t; s += t; ss += t * t;
  }
#pragma unroll
  for (int m = 16; m >= 1; m >>= 1) {
    s  += __shfl_xor(s, m, 32);
    ss += __shfl_xor(ss, m, 32);
  }
  float mu  = s * (1.f / 192.f);
  float var = ss * (1.f / 192.f) - mu * mu;
  float inv = rsqrtf(var + 1e-5f);
  bf16* dst = y + tok * C_;
#pragma unroll
  for (int j = 0; j < 6; ++j) {
    int ch = j * 32 + lane;
    dst[ch] = (bf16)((vals[j] - mu) * inv * g[ch] + be[ch]);
  }
}

// ---------------------------------------------------------------------------
// K6: fused MLP. 32 tokens per block; fc1 runs 16x64 strips per wave with
// bias+exact-GELU into a 48KB LDS bf16 intermediate; fc2 (K=768, LDS A)
// + bias + residual RMW into d_out.
// ---------------------------------------------------------------------------
__global__ __launch_bounds__(256) void k_mlp(const bf16* __restrict__ y,
                                             const bf16* __restrict__ w1,
                                             const float* __restrict__ b1,
                                             const bf16* __restrict__ w2,
                                             const float* __restrict__ b2,
                                             float* __restrict__ hio) {
  __shared__ bf16 m1[32 * HID];   // 48 KB
  long t0 = (long)blockIdx.x * 32;
  int  wave = threadIdx.x >> 5, lane = threadIdx.x & 31;
  int  half = lane >> 4, c16 = lane & 15;

  __builtin_prefetch(y + (t0 + 32) * C_, 0, 0);   // next token block

  // fc1: 2 mt x 12 n-groups of 64 -> 24 jobs, 3 per wave; 4 WMMAs per A load
  for (int tt = wave; tt < 24; tt += 8) {
    int mt = tt / 12, ng = tt % 12;
    int m0 = mt * 16, n0 = ng * 64;
    const bf16* arow = y  + (t0 + m0) * C_;
    const bf16* brow = w1 + (long)n0 * C_;
    v8f acc[4] = {};
#pragma unroll
    for (int kk = 0; kk < C_; kk += 32) {
      v16bf a = load_a(arow + kk, C_);
#pragma unroll
      for (int s = 0; s < 4; ++s)
        acc[s] = WMMA_BF16(a, load_a(brow + (long)s * 16 * C_ + kk, C_), acc[s]);
    }
#pragma unroll
    for (int s = 0; s < 4; ++s) {
      int   col = n0 + s * 16 + c16;
      float bias = b1[col];
#pragma unroll
      for (int i = 0; i < 8; ++i) {
        float xv  = acc[s][i] + bias;
        float gel = 0.5f * xv * (1.f + erff(xv * 0.70710678f));   // exact GELU
        m1[(m0 + i + (half ? 8 : 0)) * HID + col] = (bf16)gel;
      }
    }
  }
  __syncthreads();

  // fc2: 2 x 12 tiles of 16x16, K = 768 (24 WMMAs each); A from LDS
  for (int tt = wave; tt < 24; tt += 8) {
    int mt = tt / 12, nt = tt % 12;
    int m0 = mt * 16, n0 = nt * 16;
    v8f acc = {};
    const bf16* arow = &m1[m0 * HID];
    const bf16* brow = w2 + (long)n0 * HID;
#pragma unroll 4
    for (int kk = 0; kk < HID; kk += 32)
      acc = WMMA_BF16(load_a(arow + kk, HID), load_a(brow + kk, HID), acc);
    int   col = n0 + c16;
    float bias = b2[col];
#pragma unroll
    for (int i = 0; i < 8; ++i) {
      long tok = t0 + m0 + i + (half ? 8 : 0);
      hio[tok * C_ + col] += acc[i] + bias;   // h + mlp(h)
    }
  }
}

// ---------------------------------------------------------------------------
extern "C" void kernel_launch(void* const* d_in, const int* in_sizes, int n_in,
                              void* d_out, int out_size, void* d_ws, size_t ws_size,
                              hipStream_t stream) {
  (void)in_sizes; (void)n_in; (void)out_size; (void)ws_size;
  const float* x    = (const float*)d_in[0];
  const float* n1g  = (const float*)d_in[1];
  const float* n1b  = (const float*)d_in[2];
  const float* qkvw = (const float*)d_in[3];
  const float* qkvb = (const float*)d_in[4];
  const float* rpb  = (const float*)d_in[5];
  const float* pw   = (const float*)d_in[6];
  const float* pb   = (const float*)d_in[7];
  const float* n2g  = (const float*)d_in[8];
  const float* n2b  = (const float*)d_in[9];
  const float* w1   = (const float*)d_in[10];
  const float* b1   = (const float*)d_in[11];
  const float* w2   = (const float*)d_in[12];
  const float* b2   = (const float*)d_in[13];
  float* out = (float*)d_out;

  size_t e = (size_t)T * C_;               // 25,165,824 elements
  bf16* bufA = (bf16*)d_ws;                // xw, then attention output
  bf16* bufQ = bufA + e;                   // q, then ln2 output
  bf16* bufK = bufQ + e;
  bf16* bufV = bufK + e;
  bf16* wqb  = bufV + e;                   // bf16 weights
  bf16* wpb  = wqb + 576 * 192;
  bf16* w1b  = wpb + 192 * 192;
  bf16* w2b  = w1b + 768 * 192;

  k_cvt <<<576,             256, 0, stream>>>(qkvw, pw, w1, w2, wqb, wpb, w1b, w2b);
  k_ln1 <<<(int)(T / 8),    256, 0, stream>>>(x, n1g, n1b, bufA);
  k_qkv <<<(8192 * 9) / 8,  256, 0, stream>>>(bufA, wqb, qkvb, bufQ, bufK, bufV);
  k_attn<<<BW * NH_,        128, 0, stream>>>(bufQ, bufK, bufV, rpb, bufA);
  k_proj<<<(8192 * 3) / 8,  256, 0, stream>>>(bufA, wpb, pb, x, out);
  k_ln2 <<<(int)(T / 8),    256, 0, stream>>>(out, n2g, n2b, bufQ);
  k_mlp <<<(int)(T / 32),   256, 0, stream>>>(bufQ, w1b, b1, w2b, b2, out);
}
